// MultiHeadAttention_20538533609892
// MI455X (gfx1250) — matrix operand order repaired and checked by reading, not compiled, not run on hardware
//
#include <hip/hip_runtime.h>
#include <hip/hip_bf16.h>

#define D_MODEL   1024
#define NUM_HEADS 16
#define DK        64
#define BATCH     2
#define SEQ       2048
#define M_TOT     (BATCH * SEQ)   // 4096

typedef __bf16 bf16_t;
typedef __attribute__((ext_vector_type(16))) __bf16        v16bf;
typedef __attribute__((ext_vector_type(8)))  float         v8f;
typedef __attribute__((ext_vector_type(4)))  unsigned int  v4u;

#define LOG2E 1.4426950408889634f

union Frag {
    v16bf v;
    v4u   u[2];
};

__device__ __forceinline__ v4u ld128(const bf16_t* p) {
    return *reinterpret_cast<const v4u*>(p);
}

// hardware base-2 exponential: v_exp_f32
__device__ __forceinline__ float fast_exp2(float x) {
    return __builtin_amdgcn_exp2f(x);
}

// async copy 16B global -> LDS (per lane), tracked by ASYNCcnt
__device__ __forceinline__ void async_cp16(unsigned int lds_off, const bf16_t* g) {
    asm volatile("global_load_async_to_lds_b128 %0, %1, off"
                 :: "v"(lds_off), "v"((unsigned long long)(size_t)g)
                 : "memory");
}
__device__ __forceinline__ void wait_async0() {
    asm volatile("s_wait_asynccnt 0x0" ::: "memory");
}

// ---------------------------------------------------------------------------
// fp32 -> bf16 elementwise conversion
// ---------------------------------------------------------------------------
__global__ void cvt_f32_bf16(const float* __restrict__ in,
                             bf16_t* __restrict__ out, int n) {
    int i = blockIdx.x * blockDim.x + threadIdx.x;
    int stride = gridDim.x * blockDim.x;
    for (; i < n; i += stride) out[i] = (bf16_t)in[i];
}

// ---------------------------------------------------------------------------
// GEMM: Y[4096,1024] = X(bf16) @ W^T + bias.  One wave = 64x64 tile.
// MODE 0: bf16 -> [B,H,S,DK]   MODE 1: bf16 -> [B,H,DK,S]   MODE 2: f32 -> [M,N]
// ---------------------------------------------------------------------------
template <int MODE>
__global__ void __launch_bounds__(128)
gemm_bf16(const bf16_t* __restrict__ A,
          const bf16_t* __restrict__ W,
          const float*  __restrict__ bias,
          void* __restrict__ outp) {
    const int lane = threadIdx.x & 31;
    const int wave = threadIdx.x >> 5;
    const int tile = blockIdx.x * 4 + wave;        // 0..1023
    const int tile_m = tile >> 4;                  // 64 M-tiles
    const int tile_n = tile & 15;                  // 16 N-tiles
    const int m0 = tile_m * 64, n0 = tile_n * 64;
    const int lm = lane & 15;
    const bool hi = lane >= 16;

    v8f acc[4][4] = {};

    for (int k0 = 0; k0 < D_MODEL; k0 += 32) {
        Frag a[4], b[4];
#pragma unroll
        for (int mi = 0; mi < 4; mi++) {
            const bf16_t* pa = A + (size_t)(m0 + 16 * mi + lm) * D_MODEL + k0
                                 + (hi ? 8 : 0);
            a[mi].u[0] = ld128(pa);
            a[mi].u[1] = ld128(pa + 16);
        }
#pragma unroll
        for (int ni = 0; ni < 4; ni++) {
            const bf16_t* pw = W + (size_t)(n0 + 16 * ni + lm) * D_MODEL + k0
                                 + (hi ? 8 : 0);
            b[ni].u[0] = ld128(pw);
            b[ni].u[1] = ld128(pw + 16);
        }
#pragma unroll
        for (int mi = 0; mi < 4; mi++)
#pragma unroll
            for (int ni = 0; ni < 4; ni++)
                acc[mi][ni] = __builtin_amdgcn_wmma_f32_16x16x32_bf16(
                    false, a[mi].v, false, b[ni].v, (short)0, acc[mi][ni],
                    false, false);
    }

#pragma unroll
    for (int ni = 0; ni < 4; ni++) {
        const int n = n0 + 16 * ni + lm;
        const float bv = bias[n];
#pragma unroll
        for (int mi = 0; mi < 4; mi++) {
#pragma unroll
            for (int r = 0; r < 8; r++) {
                const int m = m0 + 16 * mi + r + (hi ? 8 : 0);
                const float vv = acc[mi][ni][r] + bv;
                if (MODE == 2) {
                    ((float*)outp)[(size_t)m * D_MODEL + n] = vv;
                } else {
                    const int bb = m >> 11, s = m & (SEQ - 1);
                    const int hh = n >> 6, dk = n & (DK - 1);
                    bf16_t* o = (bf16_t*)outp;
                    const size_t idx =
                        (MODE == 0)
                            ? ((size_t)(bb * NUM_HEADS + hh) * SEQ + s) * DK + dk
                            : ((size_t)(bb * NUM_HEADS + hh) * DK + dk) * SEQ + s;
                    o[idx] = (bf16_t)vv;
                }
            }
        }
    }
}

// ---------------------------------------------------------------------------
// Flash attention.
//  - block = 4 waves, one (b,h), 128 queries; wave owns 32 queries (2 col-tiles)
//  - K/V tile (32 keys) staged in double-buffered LDS via async-to-LDS loads,
//    shared by all 4 waves; next tile's loads overlap current compute.
//  - scores computed transposed: S_t[key,q] = K_tile @ Q^T; softmax over keys
//    is 16 in-lane values + one shfl_xor(16); O_t[d,q] += V^T @ P_t.
// Qp,Kp: bf16 [B,H,S,64]; Vt: bf16 [B,H,64,S]; X: bf16 [4096,1024]
// ---------------------------------------------------------------------------
#define KPAD 72   // K tile row pitch (elements): 144B, 16B-multiple, spreads banks
#define VPAD 40   // V tile row pitch (elements):  80B, 16B-multiple

__global__ void __launch_bounds__(128)
flash_attn(const bf16_t* __restrict__ Qp,
           const bf16_t* __restrict__ Kp,
           const bf16_t* __restrict__ Vt,
           bf16_t* __restrict__ X) {
    __shared__ bf16_t sK[2][32][KPAD];   // [buf][key][d]
    __shared__ bf16_t sV[2][64][VPAD];   // [buf][d][key]

    const int tid  = threadIdx.x;
    const int lane = tid & 31;
    const int wave = tid >> 5;
    const int bh   = blockIdx.x >> 4;    // 32 heads-total, 16 blocks each
    const int qblk = blockIdx.x & 15;
    const int q0   = qblk * 128 + wave * 32;   // this wave: queries q0..q0+31
    const int lm   = lane & 15;
    const bool hi  = lane >= 16;

    const bf16_t* Qh = Qp + (size_t)bh * SEQ * DK;
    const bf16_t* Kh = Kp + (size_t)bh * SEQ * DK;
    const bf16_t* Vh = Vt + (size_t)bh * DK * SEQ;

    // ---- staging map: 4 async 16B chunks per thread per tile ----
    const int krow = tid >> 3;           // 0..15  (plus +16 row)
    const int kcol = (tid & 7) * 8;      // element offset 0..56
    const int vrow = tid >> 2;           // 0..31  (plus +32 row)
    const int vcol = (tid & 3) * 8;      // element offset 0..24
    unsigned int lK1[2], lK2[2], lV1[2], lV2[2];
#pragma unroll
    for (int b = 0; b < 2; b++) {
        lK1[b] = (unsigned int)(size_t)&sK[b][krow][kcol];
        lK2[b] = (unsigned int)(size_t)&sK[b][krow + 16][kcol];
        lV1[b] = (unsigned int)(size_t)&sV[b][vrow][vcol];
        lV2[b] = (unsigned int)(size_t)&sV[b][vrow + 32][vcol];
    }
    auto stage = [&](int b, int k0) {
        const bf16_t* gK = Kh + (size_t)(k0 + krow) * DK + kcol;
        const bf16_t* gV = Vh + (size_t)vrow * SEQ + k0 + vcol;
        async_cp16(lK1[b], gK);
        async_cp16(lK2[b], gK + 16 * DK);
        async_cp16(lV1[b], gV);
        async_cp16(lV2[b], gV + 32 * (size_t)SEQ);
    };

    // ---- Q fragments: 2 query tiles x 2 d-halves (B-matrix layout) ----
    Frag qf[2][2];
#pragma unroll
    for (int qt = 0; qt < 2; qt++)
#pragma unroll
        for (int j = 0; j < 2; j++) {
            const bf16_t* pq = Qh + (size_t)(q0 + 16 * qt + lm) * DK + 32 * j
                                  + (hi ? 8 : 0);
            qf[qt][j].u[0] = ld128(pq);
            qf[qt][j].u[1] = ld128(pq + 16);
        }

    v8f od[2][4] = {};
    float m2[2] = {-__builtin_inff(), -__builtin_inff()};
    float lr[2] = {0.0f, 0.0f};
    const float SC2 = 0.125f * LOG2E;    // softmax in base-2 domain

    stage(0, 0);

    for (int it = 0; it < SEQ / 32; ++it) {
        const int b = it & 1;
        wait_async0();
        __syncthreads();
        if (it + 1 < SEQ / 32) stage(b ^ 1, (it + 1) * 32);

        // --- K fragments from LDS (A-matrix: 16 keys x 32 d) ---
        Frag kf0[2], kf1[2];
#pragma unroll
        for (int j = 0; j < 2; j++) {
            const bf16_t* p0 = &sK[b][lm][32 * j + (hi ? 8 : 0)];
            kf0[j].u[0] = ld128(p0);
            kf0[j].u[1] = ld128(p0 + 16);
            const bf16_t* p1 = &sK[b][16 + lm][32 * j + (hi ? 8 : 0)];
            kf1[j].u[0] = ld128(p1);
            kf1[j].u[1] = ld128(p1 + 16);
        }

#pragma unroll
        for (int qt = 0; qt < 2; qt++) {
            v8f s0 = {}, s1 = {};
            s0 = __builtin_amdgcn_wmma_f32_16x16x32_bf16(
                false, kf0[0].v, false, qf[qt][0].v, (short)0, s0, false, false);
            s0 = __builtin_amdgcn_wmma_f32_16x16x32_bf16(
                false, kf0[1].v, false, qf[qt][1].v, (short)0, s0, false, false);
            s1 = __builtin_amdgcn_wmma_f32_16x16x32_bf16(
                false, kf1[0].v, false, qf[qt][0].v, (short)0, s1, false, false);
            s1 = __builtin_amdgcn_wmma_f32_16x16x32_bf16(
                false, kf1[1].v, false, qf[qt][1].v, (short)0, s1, false, false);

            // --- online softmax (base 2), per query = per lane ---
            float p0[8], p1[8];
#pragma unroll
            for (int r = 0; r < 8; r++) { p0[r] = s0[r] * SC2; p1[r] = s1[r] * SC2; }
            float mx = p0[0];
#pragma unroll
            for (int r = 1; r < 8; r++) mx = fmaxf(mx, p0[r]);
#pragma unroll
            for (int r = 0; r < 8; r++) mx = fmaxf(mx, p1[r]);
            mx = fmaxf(mx, __shfl_xor(mx, 16, 32));
            const float mnew = fmaxf(m2[qt], mx);
            const float alpha = fast_exp2(m2[qt] - mnew);
            float ls = 0.0f;
#pragma unroll
            for (int r = 0; r < 8; r++) { p0[r] = fast_exp2(p0[r] - mnew); ls += p0[r]; }
#pragma unroll
            for (int r = 0; r < 8; r++) { p1[r] = fast_exp2(p1[r] - mnew); ls += p1[r]; }
            ls += __shfl_xor(ls, 16, 32);
            lr[qt] = lr[qt] * alpha + ls;
            m2[qt] = mnew;
#pragma unroll
            for (int t = 0; t < 4; t++)
#pragma unroll
                for (int r = 0; r < 8; r++) od[qt][t][r] *= alpha;

            // --- rebuild P_t as B-matrix fragment (keys x queries) ---
            Frag pb;
#pragma unroll
            for (int r = 0; r < 8; r++) {
                const float o0 = __shfl_xor(p0[r], 16, 32);
                const float o1 = __shfl_xor(p1[r], 16, 32);
                const float lo = hi ? o1 : p0[r];   // keys {0..7}  / {16..23}
                const float hv = hi ? p1[r] : o0;   // keys {8..15} / {24..31}
                pb.v[r]     = (bf16_t)lo;
                pb.v[8 + r] = (bf16_t)hv;
            }

            // --- O_t += V^T @ P_t (A fragments from LDS V^T tile) ---
#pragma unroll
            for (int t = 0; t < 4; t++) {
                Frag vf;
                const bf16_t* pv = &sV[b][16 * t + lm][hi ? 16 : 0];
                vf.u[0] = ld128(pv);
                vf.u[1] = ld128(pv + 8);
                od[qt][t] = __builtin_amdgcn_wmma_f32_16x16x32_bf16(
                    false, vf.v, false, pb.v, (short)0, od[qt][t], false, false);
            }
        }
    }

    const int h = bh & (NUM_HEADS - 1);
#pragma unroll
    for (int qt = 0; qt < 2; qt++) {
        const float inv = 1.0f / lr[qt];
        const int m_token = (bh >> 4) * SEQ + q0 + 16 * qt + lm;  // b*S + s
#pragma unroll
        for (int t = 0; t < 4; t++)
#pragma unroll
            for (int r = 0; r < 8; r++) {
                const int col = h * DK + 16 * t + r + (hi ? 8 : 0);
                X[(size_t)m_token * D_MODEL + col] = (bf16_t)(od[qt][t][r] * inv);
            }
    }
}

// ---------------------------------------------------------------------------
extern "C" void kernel_launch(void* const* d_in, const int* in_sizes, int n_in,
                              void* d_out, int out_size, void* d_ws,
                              size_t ws_size, hipStream_t stream) {
    (void)in_sizes; (void)n_in; (void)out_size; (void)ws_size;

    const float* q   = (const float*)d_in[0];
    const float* k   = (const float*)d_in[1];
    const float* v   = (const float*)d_in[2];
    const float* w_q = (const float*)d_in[3];
    const float* b_q = (const float*)d_in[4];
    const float* w_k = (const float*)d_in[5];
    const float* b_k = (const float*)d_in[6];
    const float* w_v = (const float*)d_in[7];
    const float* b_v = (const float*)d_in[8];
    const float* w_o = (const float*)d_in[9];
    const float* b_o = (const float*)d_in[10];

    char* ws = (char*)d_ws;
    size_t off = 0;
    auto alloc = [&](size_t bytes) {
        void* p = ws + off;
        off += (bytes + 255) & ~(size_t)255;
        return p;
    };
    const size_t MD = (size_t)M_TOT * D_MODEL;
    const size_t DD = (size_t)D_MODEL * D_MODEL;

    bf16_t* qb  = (bf16_t*)alloc(MD * 2);
    bf16_t* kb  = (bf16_t*)alloc(MD * 2);
    bf16_t* vb  = (bf16_t*)alloc(MD * 2);
    bf16_t* wqb = (bf16_t*)alloc(DD * 2);
    bf16_t* wkb = (bf16_t*)alloc(DD * 2);
    bf16_t* wvb = (bf16_t*)alloc(DD * 2);
    bf16_t* wob = (bf16_t*)alloc(DD * 2);
    bf16_t* Qp  = (bf16_t*)alloc(MD * 2);  // [B,H,S,DK]
    bf16_t* Kp  = (bf16_t*)alloc(MD * 2);  // [B,H,S,DK]
    bf16_t* Vt  = (bf16_t*)alloc(MD * 2);  // [B,H,DK,S]
    bf16_t* Xa  = (bf16_t*)alloc(MD * 2);  // attention output

    cvt_f32_bf16<<<2048, 256, 0, stream>>>(q, qb, (int)MD);
    cvt_f32_bf16<<<2048, 256, 0, stream>>>(k, kb, (int)MD);
    cvt_f32_bf16<<<2048, 256, 0, stream>>>(v, vb, (int)MD);
    cvt_f32_bf16<<<512, 256, 0, stream>>>(w_q, wqb, (int)DD);
    cvt_f32_bf16<<<512, 256, 0, stream>>>(w_k, wkb, (int)DD);
    cvt_f32_bf16<<<512, 256, 0, stream>>>(w_v, wvb, (int)DD);
    cvt_f32_bf16<<<512, 256, 0, stream>>>(w_o, wob, (int)DD);

    gemm_bf16<0><<<256, 128, 0, stream>>>(qb, wqb, b_q, (void*)Qp);
    gemm_bf16<0><<<256, 128, 0, stream>>>(kb, wkb, b_k, (void*)Kp);
    gemm_bf16<1><<<256, 128, 0, stream>>>(vb, wvb, b_v, (void*)Vt);

    flash_attn<<<512, 128, 0, stream>>>(Qp, Kp, Vt, Xa);

    gemm_bf16<2><<<256, 128, 0, stream>>>(Xa, wob, b_o, d_out);
}
